// MyModel_30399778521295
// MI455X (gfx1250) — compile-verified, matrix operand order checked by reading, not run on hardware
//
#include <hip/hip_runtime.h>
#include <hip/hip_bf16.h>

// ---------------------------------------------------------------------------
// BiLSTM-CRF NLL on gfx1250 (MI455X).
// Matrix work uses v_wmma_f32_16x16x32_bf16 (wave32 WMMA, f32 accumulate).
// ---------------------------------------------------------------------------

typedef __attribute__((ext_vector_type(16))) __bf16 v16bf;
typedef __attribute__((ext_vector_type(8)))  __bf16 v8bf;
typedef __attribute__((ext_vector_type(8)))  float  v8f;

#define E_    768
#define H_    128
#define G4H   512      // 4*H (gate width)
#define B_    64
#define S_    512
#define T_    9
#define KS_IH 24       // 768/32 k-steps for input GEMM
#define NT4H  32       // 512/16 N tiles over gates
#define MT    2048     // M-tiles for input GEMM: 4 batch-tiles * 512 timesteps

// K index inside a 16-bit A/B fragment (ISA 7.12.2): element e of a lane.
__device__ __forceinline__ int k_of(int lane, int e) {
    return (e < 8 ? e : e + 8) + ((lane & 16) ? 8 : 0);
}

// ---------------------------------------------------------------------------
// K1: pack Wih/Whh into WMMA B-fragment layouts, combine biases.
//   wih_pk[dir][nt(32)][ks(24)][lane(32)][e(16)]   (bf16)
//   whh_pk[dir][wave(8)][gate(4)][ks(4)][lane(32)][e(16)] (bf16)
//   bias_c[dir][512] = bih + bhh (f32)
// ---------------------------------------------------------------------------
__global__ void pack_kernel(const float* __restrict__ Wih_f, const float* __restrict__ Whh_f,
                            const float* __restrict__ bih_f, const float* __restrict__ bhh_f,
                            const float* __restrict__ Wih_b, const float* __restrict__ Whh_b,
                            const float* __restrict__ bih_b, const float* __restrict__ bhh_b,
                            __bf16* __restrict__ wih_pk, __bf16* __restrict__ whh_pk,
                            float* __restrict__ bias_c) {
    const int N_WIH = 2 * NT4H * KS_IH * 32;   // 49152 lane-slots
    const int N_WHH = 2 * 8 * 4 * 4 * 32;      // 8192 lane-slots
    int idx = blockIdx.x * 256 + threadIdx.x;
    if (idx < N_WIH) {
        int lane = idx & 31; int r = idx >> 5;
        int ks = r % KS_IH; r /= KS_IH;
        int nt = r % NT4H;  int dir = r / NT4H;
        const float* W = dir ? Wih_b : Wih_f;
        int n = nt * 16 + (lane & 15);
        __bf16* dst = wih_pk + (size_t)idx * 16;
        for (int e = 0; e < 16; ++e)
            dst[e] = (__bf16)W[n * E_ + ks * 32 + k_of(lane, e)];
    } else if (idx < N_WIH + N_WHH) {
        int j = idx - N_WIH;
        int lane = j & 31; int r = j >> 5;
        int ks = r & 3; r >>= 2;
        int g  = r & 3; r >>= 2;
        int w  = r & 7; int dir = r >> 3;
        const float* W = dir ? Whh_b : Whh_f;
        int n = g * 128 + w * 16 + (lane & 15);
        __bf16* dst = whh_pk + (size_t)j * 16;
        for (int e = 0; e < 16; ++e)
            dst[e] = (__bf16)W[n * H_ + ks * 32 + k_of(lane, e)];
    } else if (idx < N_WIH + N_WHH + 2 * G4H) {
        int j = idx - N_WIH - N_WHH;
        int dir = j >> 9; int n = j & 511;
        bias_c[j] = dir ? (bih_b[n] + bhh_b[n]) : (bih_f[n] + bhh_f[n]);
    }
}

// ---------------------------------------------------------------------------
// K2: f32 -> bf16 convert of bert_feats (row-major [b][s][e]).
// ---------------------------------------------------------------------------
__global__ void cvt_kernel(const float* __restrict__ x, __bf16* __restrict__ xb, int n4) {
    int i = blockIdx.x * 256 + threadIdx.x;
    if (i < n4) {
        float4 v = ((const float4*)x)[i];
        __bf16* d = xb + (size_t)i * 4;
        d[0] = (__bf16)v.x; d[1] = (__bf16)v.y; d[2] = (__bf16)v.z; d[3] = (__bf16)v.w;
    }
}

// ---------------------------------------------------------------------------
// K3: input-projection GEMM: pre = x @ Wih.T + (bih+bhh), per direction.
// M-tile = 16 batch rows at a fixed timestep (matches scan tiling).
// Output stored in raw C-fragment order:
//   pre_pk[dir][mtile(2048)][wave(8)][gate(4)][lane(32)][v(8)] (f32)
// so the scan reloads accumulators with coalesced 32B/lane loads.
// ---------------------------------------------------------------------------
__global__ void __launch_bounds__(256) igemm_kernel(const __bf16* __restrict__ xb,
                                                    const __bf16* __restrict__ wih_pk,
                                                    const float* __restrict__ bias_c,
                                                    float* __restrict__ pre_pk) {
    const int mtile = blockIdx.x;             // btile*512 + t
    const int dir   = blockIdx.y;
    const int wv    = threadIdx.x >> 5;
    const int lane  = threadIdx.x & 31;
    const int btile = mtile >> 9;
    const int t     = mtile & 511;
    const int brow  = btile * 16 + (lane & 15);
    const size_t arow = ((size_t)brow * S_ + t) * E_;
    const int koff  = (lane & 16) ? 8 : 0;

    v8f acc[4] = {};
    for (int ks = 0; ks < KS_IH; ++ks) {
        v8bf a0 = *(const v8bf*)(xb + arow + ks * 32 + koff);
        v8bf a1 = *(const v8bf*)(xb + arow + ks * 32 + koff + 16);
        v16bf a = __builtin_shufflevector(a0, a1, 0,1,2,3,4,5,6,7,8,9,10,11,12,13,14,15);
#pragma unroll
        for (int g = 0; g < 4; ++g) {
            int nt = g * 8 + wv;
            v16bf b = *(const v16bf*)(wih_pk + (((size_t)(dir * NT4H + nt) * KS_IH + ks) * 32 + lane) * 16);
            acc[g] = __builtin_amdgcn_wmma_f32_16x16x32_bf16(false, a, false, b, (short)0, acc[g], false, false);
        }
    }
#pragma unroll
    for (int g = 0; g < 4; ++g) {
        float bv = bias_c[dir * G4H + g * 128 + wv * 16 + (lane & 15)];
#pragma unroll
        for (int v = 0; v < 8; ++v) acc[g][v] += bv;
        *(v8f*)(pre_pk + ((((size_t)dir * MT + mtile) * 8 + wv) * 4 + g) * 256 + (size_t)lane * 8) = acc[g];
    }
}

// ---------------------------------------------------------------------------
// K4: persistent LSTM scan. Grid = (4 batch-tiles, 2 directions), 8 waves.
// Wave w owns hidden columns [16w,16w+16) of ALL four gates, so i/f/g/o and
// the c/h update are wave-local in identical C-fragment layout.
// Whh B-fragments live in registers (128 VGPRs); h is shared via 4KB LDS.
// ---------------------------------------------------------------------------
__global__ void __launch_bounds__(256) scan_kernel(const float* __restrict__ pre_pk,
                                                   const __bf16* __restrict__ whh_pk,
                                                   __bf16* __restrict__ hout) {
    const int btile = blockIdx.x;
    const int dir   = blockIdx.y;
    const int wv    = threadIdx.x >> 5;
    const int lane  = threadIdx.x & 31;
    const int mloc  = (lane & 16) ? 8 : 0;    // C-layout row base
    const int nloc  = lane & 15;              // C-layout column
    const int koff  = (lane & 16) ? 8 : 0;    // A-layout K offset

    __shared__ __align__(16) __bf16 hbuf[16 * H_];

    // Whh B-fragments -> registers (persist across all 512 steps)
    v16bf Bf[16];
#pragma unroll
    for (int g = 0; g < 4; ++g)
#pragma unroll
        for (int ks = 0; ks < 4; ++ks)
            Bf[g * 4 + ks] = *(const v16bf*)(whh_pk +
                ((((size_t)(dir * 8 + wv) * 4 + g) * 4 + ks) * 32 + lane) * 16);

    for (int i = threadIdx.x; i < 16 * H_; i += 256) hbuf[i] = (__bf16)0.0f;
    v8f cst = {};   // cell state: element v = (m = v+mloc, n = wv*16+nloc)
    __syncthreads();

    for (int step = 0; step < S_; ++step) {
        const int t  = dir ? (S_ - 1 - step) : step;
        const int mt = btile * S_ + t;

        // C init = precomputed input projection (fragment-order, coalesced)
        v8f acc[4];
#pragma unroll
        for (int g = 0; g < 4; ++g)
            acc[g] = *(const v8f*)(pre_pk + ((((size_t)dir * MT + mt) * 8 + wv) * 4 + g) * 256 + (size_t)lane * 8);
        if (step + 1 < S_) {
            const int mtn = btile * S_ + (dir ? (t - 1) : (t + 1));
            __builtin_prefetch(pre_pk + (((size_t)dir * MT + mtn) * 8 + wv) * 1024 + (size_t)lane * 8, 0, 0);
        }

        // A fragments = h(t-1) from LDS
        v16bf Af[4];
        const int m = lane & 15;
#pragma unroll
        for (int ks = 0; ks < 4; ++ks) {
            v8bf a0 = *(const v8bf*)(hbuf + m * H_ + ks * 32 + koff);
            v8bf a1 = *(const v8bf*)(hbuf + m * H_ + ks * 32 + koff + 16);
            Af[ks] = __builtin_shufflevector(a0, a1, 0,1,2,3,4,5,6,7,8,9,10,11,12,13,14,15);
        }
        __syncthreads();   // all waves consumed h(t-1); safe to overwrite

        // gates += h @ Whh.T : 16 WMMAs per wave
#pragma unroll
        for (int g = 0; g < 4; ++g)
#pragma unroll
            for (int ks = 0; ks < 4; ++ks)
                acc[g] = __builtin_amdgcn_wmma_f32_16x16x32_bf16(false, Af[ks], false, Bf[g * 4 + ks],
                                                                 (short)0, acc[g], false, false);

        // LSTM cell (wave-local, identical layouts across the four gate tiles)
#pragma unroll
        for (int v = 0; v < 8; ++v) {
            float iv = 1.0f / (1.0f + __expf(-acc[0][v]));
            float fv = 1.0f / (1.0f + __expf(-acc[1][v]));
            float gv = tanhf(acc[2][v]);
            float ov = 1.0f / (1.0f + __expf(-acc[3][v]));
            float cv = fv * cst[v] + iv * gv;
            cst[v] = cv;
            float hv = ov * tanhf(cv);
            __bf16 hb = (__bf16)hv;
            int mr = v + mloc;
            hbuf[mr * H_ + wv * 16 + nloc] = hb;
            hout[(((size_t)dir * B_ + btile * 16 + mr) * S_ + t) * H_ + wv * 16 + nloc] = hb;
        }
        __syncthreads();
    }
}

// ---------------------------------------------------------------------------
// K5: emissions = concat(h_f, h_b) @ Wt.T + bt   (tiny: 150 MFLOP)
// ---------------------------------------------------------------------------
__global__ void emis_kernel(const __bf16* __restrict__ hout, const float* __restrict__ Wt,
                            const float* __restrict__ bt, float* __restrict__ em) {
    int idx = blockIdx.x * 256 + threadIdx.x;
    if (idx >= B_ * S_ * T_) return;
    int tag = idx % T_;
    int token = idx / T_;           // b*512 + t
    float acc = bt[tag];
    const __bf16* hf = hout + (size_t)token * H_;
    const __bf16* hb = hout + ((size_t)B_ * S_ + token) * H_;
    const float* w = Wt + tag * 2 * H_;
    for (int k = 0; k < H_; ++k) acc += (float)hf[k] * w[k];
    for (int k = 0; k < H_; ++k) acc += (float)hb[k] * w[H_ + k];
    em[idx] = acc;
}

// ---------------------------------------------------------------------------
// K6: CRF numerator (gold path score), one thread per batch element.
// ---------------------------------------------------------------------------
__global__ void num_kernel(const float* __restrict__ em, const int* __restrict__ tags,
                           const int* __restrict__ lengths, const float* __restrict__ start_t,
                           const float* __restrict__ end_t, const float* __restrict__ trans,
                           float* __restrict__ nums) {
    int b = threadIdx.x;
    if (b >= B_) return;
    int len = lengths[b]; len = len < 1 ? 1 : (len > S_ ? S_ : len);
    const int* tg = tags + b * S_;
    float acc = start_t[tg[0]] + em[(b * S_) * T_ + tg[0]];
    for (int t = 1; t < len; ++t)
        acc += trans[tg[t - 1] * T_ + tg[t]] + em[(b * S_ + t) * T_ + tg[t]];
    acc += end_t[tg[len - 1]];
    nums[b] = acc;
}

// ---------------------------------------------------------------------------
// K7: CRF denominator (forward algorithm), one wave per batch element;
// lanes 0..8 each own one tag's alpha recursion.
// ---------------------------------------------------------------------------
__global__ void den_kernel(const float* __restrict__ em, const int* __restrict__ lengths,
                           const float* __restrict__ start_t, const float* __restrict__ end_t,
                           const float* __restrict__ trans, float* __restrict__ dens) {
    int b = blockIdx.x;
    int j = threadIdx.x;
    __shared__ float alpha[T_];
    __shared__ float tr_s[T_ * T_];
    for (int i = j; i < T_ * T_; i += 32) tr_s[i] = trans[i];
    if (j < T_) alpha[j] = start_t[j] + em[(b * S_) * T_ + j];
    int len = lengths[b]; len = len < 1 ? 1 : (len > S_ ? S_ : len);
    __syncthreads();
    for (int t = 1; t < len; ++t) {
        float nxt = 0.0f;
        if (j < T_) {
            float mx = -1e30f;
            for (int i = 0; i < T_; ++i) mx = fmaxf(mx, alpha[i] + tr_s[i * T_ + j]);
            float s = 0.0f;
            for (int i = 0; i < T_; ++i) s += __expf(alpha[i] + tr_s[i * T_ + j] - mx);
            nxt = mx + __logf(s) + em[(b * S_ + t) * T_ + j];
        }
        __syncthreads();
        if (j < T_) alpha[j] = nxt;
        __syncthreads();
    }
    if (j == 0) {
        float mx = -1e30f;
        for (int i = 0; i < T_; ++i) mx = fmaxf(mx, alpha[i] + end_t[i]);
        float s = 0.0f;
        for (int i = 0; i < T_; ++i) s += __expf(alpha[i] + end_t[i] - mx);
        dens[b] = mx + __logf(s);
    }
}

// K8: out = -mean(num - den)
__global__ void final_kernel(const float* __restrict__ nums, const float* __restrict__ dens,
                             float* __restrict__ out) {
    if (threadIdx.x == 0) {
        float s = 0.0f;
        for (int b = 0; b < B_; ++b) s += nums[b] - dens[b];
        out[0] = -s / (float)B_;
    }
}

// ---------------------------------------------------------------------------
extern "C" void kernel_launch(void* const* d_in, const int* in_sizes, int n_in,
                              void* d_out, int out_size, void* d_ws, size_t ws_size,
                              hipStream_t stream) {
    const float* x       = (const float*)d_in[0];
    const int*   tags    = (const int*)  d_in[1];
    const int*   lengths = (const int*)  d_in[2];
    const float* Wih_f   = (const float*)d_in[3];
    const float* Whh_f   = (const float*)d_in[4];
    const float* bih_f   = (const float*)d_in[5];
    const float* bhh_f   = (const float*)d_in[6];
    const float* Wih_b   = (const float*)d_in[7];
    const float* Whh_b   = (const float*)d_in[8];
    const float* bih_b   = (const float*)d_in[9];
    const float* bhh_b   = (const float*)d_in[10];
    const float* Wt      = (const float*)d_in[11];
    const float* bt      = (const float*)d_in[12];
    const float* start_t = (const float*)d_in[13];
    const float* end_t   = (const float*)d_in[14];
    const float* trans   = (const float*)d_in[15];
    float* out = (float*)d_out;

    char* ws = (char*)d_ws;
    size_t off = 0;
    auto carve = [&](size_t bytes) -> void* {
        void* p = ws + off;
        off = (off + bytes + 255) & ~(size_t)255;
        return p;
    };
    __bf16* xb      = (__bf16*)carve((size_t)B_ * S_ * E_ * 2);                  // 50.3 MB
    __bf16* wih_pk  = (__bf16*)carve((size_t)2 * NT4H * KS_IH * 32 * 16 * 2);    // 1.5 MB
    __bf16* whh_pk  = (__bf16*)carve((size_t)2 * 8 * 4 * 4 * 32 * 16 * 2);       // 256 KB
    float*  bias_c  = (float*) carve((size_t)2 * G4H * 4);
    float*  pre_pk  = (float*) carve((size_t)2 * MT * 8 * 4 * 256 * 4);          // 128 MB
    __bf16* hout    = (__bf16*)carve((size_t)2 * B_ * S_ * H_ * 2);              // 16 MB
    float*  em      = (float*) carve((size_t)B_ * S_ * T_ * 4);                  // 1.2 MB
    float*  nums    = (float*) carve((size_t)B_ * 4);
    float*  dens    = (float*) carve((size_t)B_ * 4);
    (void)ws_size; (void)in_sizes; (void)n_in; (void)out_size;

    pack_kernel<<<228, 256, 0, stream>>>(Wih_f, Whh_f, bih_f, bhh_f,
                                         Wih_b, Whh_b, bih_b, bhh_b,
                                         wih_pk, whh_pk, bias_c);
    const int n4 = B_ * S_ * E_ / 4;
    cvt_kernel<<<(n4 + 255) / 256, 256, 0, stream>>>(x, xb, n4);
    igemm_kernel<<<dim3(MT, 2), 256, 0, stream>>>(xb, wih_pk, bias_c, pre_pk);
    scan_kernel<<<dim3(4, 2), 256, 0, stream>>>(pre_pk, whh_pk, hout);
    emis_kernel<<<(B_ * S_ * T_ + 255) / 256, 256, 0, stream>>>(hout, Wt, bt, em);
    num_kernel<<<1, 64, 0, stream>>>(em, tags, lengths, start_t, end_t, trans, nums);
    den_kernel<<<B_, 32, 0, stream>>>(em, lengths, start_t, end_t, trans, dens);
    final_kernel<<<1, 32, 0, stream>>>(nums, dens, out);
}
